// Head_17025250361437
// MI455X (gfx1250) — matrix operand order repaired and checked by reading, not compiled, not run on hardware
//
#include <hip/hip_runtime.h>

#define Bz 16
#define Tz 2048
#define Dz 1024
#define Hz 1024

typedef __attribute__((ext_vector_type(16))) __bf16 v16bf;
typedef __attribute__((ext_vector_type(8)))  float  v8f;
typedef __attribute__((ext_vector_type(4)))  int    v4i;
typedef __attribute__((ext_vector_type(4)))  float  v4f;

#define WMMA_BF16(a, b, c) \
  __builtin_amdgcn_wmma_f32_16x16x32_bf16(false, (a), false, (b), (short)0, (c), false, false)

// Fill a 16x32 (A) or 32x16 (B) bf16 fragment with two 128-bit loads.
__device__ __forceinline__ v16bf frag_ld(const __bf16* p0, const __bf16* p1) {
  union { v16bf v; v4i d[2]; } u;
  u.d[0] = *(const v4i*)p0;
  u.d[1] = *(const v4i*)p1;
  return u.v;
}

// Async DMA one 16-byte chunk global -> LDS (ASYNCcnt-tracked, no VGPR data).
__device__ __forceinline__ void async_ld_b128(unsigned lds_addr, unsigned gvoff,
                                              unsigned long long sbase) {
  asm volatile("global_load_async_to_lds_b128 %0, %1, %2"
               :: "v"(lds_addr), "v"(gvoff), "s"(sbase) : "memory");
}

// ---------------------------------------------------------------------------
// Prep 1: x fp32 -> xb bf16 (packed b128 in / b128 out), 8 elems per thread.
// ---------------------------------------------------------------------------
__global__ __launch_bounds__(256) void convert_x_kernel(const float* __restrict__ x,
                                                        __bf16* __restrict__ xb) {
  const size_t i0 = ((size_t)blockIdx.x * 256 + threadIdx.x) * 8;
  v4f f0 = *(const v4f*)(x + i0);
  v4f f1 = *(const v4f*)(x + i0 + 4);
  union { __bf16 h[8]; v4i d; } u;
#pragma unroll
  for (int j = 0; j < 4; ++j) u.h[j]     = (__bf16)f0[j];
#pragma unroll
  for (int j = 0; j < 4; ++j) u.h[4 + j] = (__bf16)f1[j];
  *(v4i*)(xb + i0) = u.d;
}

// ---------------------------------------------------------------------------
// Prep 2: W fp32 [D][H] -> Wt bf16 [z][H][D] (transposed so GEMM B-tiles are
// contiguous along K). 3M elements total -- negligible cost.
// ---------------------------------------------------------------------------
__global__ __launch_bounds__(256) void transpose_w_kernel(
    const float* __restrict__ Wq, const float* __restrict__ Wk,
    const float* __restrict__ Wv, __bf16* __restrict__ Wt) {
  const int z = blockIdx.z;
  const float* W = (z == 0) ? Wq : (z == 1) ? Wk : Wv;
  const int n  = blockIdx.x * 256 + threadIdx.x;  // H index (coalesced read)
  const int kk = blockIdx.y;                       // D index
  Wt[((size_t)z * Hz + n) * Dz + kk] = (__bf16)W[(size_t)kk * Hz + n];
}

// ---------------------------------------------------------------------------
// Kernel 1: qkv = xb @ Wt^T  (all bf16, fp32 accum; V stored transposed).
//   q  : [B*T][H]  row-major bf16
//   k  : [B*T][H]  row-major bf16
//   vT : [B][H][T] (per-batch transposed) bf16
// Block: 256 threads (8 waves), tile 128(M) x 128(N), K-step 32.
// Tiles staged with double-buffered global_load_async_to_lds_b128:
// stage kt+1 DMA overlaps stage kt WMMAs.
// Waves 4(m) x 2(n); each wave computes 32x64 (2x4 WMMA frags).
// ---------------------------------------------------------------------------
__global__ __launch_bounds__(256) void qkv_gemm_kernel(
    const __bf16* __restrict__ xb,
    const __bf16* __restrict__ Wt,
    __bf16* __restrict__ qkv) {
  constexpr int ST = 40;  // LDS row stride in halves (80B: 16B-aligned, conflict-light)
  constexpr int nK = Dz / 32;
  __shared__ __bf16 As[2][128 * ST];  // x tile   [m][k]
  __shared__ __bf16 Bs[2][128 * ST];  // Wt tile  [n][k]

  const int z    = blockIdx.z;
  const int tid  = threadIdx.x;
  const int lane = tid & 31;
  const int wid  = tid >> 5;
  const int ln   = lane & 15;
  const int hl   = lane >> 4;
  const int wm   = wid >> 1;  // 0..3
  const int wn   = wid & 1;   // 0..1
  const size_t row0 = (size_t)blockIdx.y * 128;
  const int    col0 = blockIdx.x * 128;

  const unsigned long long xbase = (unsigned long long)(uintptr_t)xb;
  const unsigned long long wbase =
      (unsigned long long)(uintptr_t)(Wt + (size_t)z * Hz * Dz);

  // Each thread DMAs 2 A-chunks + 2 B-chunks (16B each) per stage.
  const int r0 = tid >> 2;        // 0..63
  const int r1 = r0 + 64;         // 64..127
  const int cc = (tid & 3) * 8;   // chunk offset within a 32-half row

  auto issue = [&](int kt, int buf) {
    const unsigned a0 = (unsigned)(((row0 + r0) * Dz + kt * 32 + cc) * 2);
    const unsigned a1 = (unsigned)(((row0 + r1) * Dz + kt * 32 + cc) * 2);
    const unsigned b0 = (unsigned)((((size_t)(col0 + r0)) * Dz + kt * 32 + cc) * 2);
    const unsigned b1 = (unsigned)((((size_t)(col0 + r1)) * Dz + kt * 32 + cc) * 2);
    async_ld_b128((unsigned)(uintptr_t)&As[buf][r0 * ST + cc], a0, xbase);
    async_ld_b128((unsigned)(uintptr_t)&As[buf][r1 * ST + cc], a1, xbase);
    async_ld_b128((unsigned)(uintptr_t)&Bs[buf][r0 * ST + cc], b0, wbase);
    async_ld_b128((unsigned)(uintptr_t)&Bs[buf][r1 * ST + cc], b1, wbase);
  };

  v8f acc[2][4] = {};

  issue(0, 0);
  for (int kt = 0; kt < nK; ++kt) {
    const int buf = kt & 1;
    if (kt + 1 < nK) {
      issue(kt + 1, buf ^ 1);                              // prefetch next stage
      asm volatile("s_wait_asynccnt 4" ::: "memory");      // stage kt complete
    } else {
      asm volatile("s_wait_asynccnt 0" ::: "memory");
    }
    __syncthreads();

    v16bf a[2], b[4];
#pragma unroll
    for (int mi = 0; mi < 2; ++mi) {
      const int r = wm * 32 + mi * 16 + ln;
      a[mi] = frag_ld(&As[buf][r * ST + 8 * hl], &As[buf][r * ST + 16 + 8 * hl]);
    }
#pragma unroll
    for (int ni = 0; ni < 4; ++ni) {
      const int n = wn * 64 + ni * 16 + ln;
      b[ni] = frag_ld(&Bs[buf][n * ST + 8 * hl], &Bs[buf][n * ST + 16 + 8 * hl]);
    }
#pragma unroll
    for (int mi = 0; mi < 2; ++mi)
#pragma unroll
      for (int ni = 0; ni < 4; ++ni)
        acc[mi][ni] = WMMA_BF16(a[mi], b[ni], acc[mi][ni]);
    __syncthreads();  // protect buf from next prefetch overwrite
  }

  // --- epilogue (branch + division hoisted; C/D layout m = j+8*hl, n = ln)
  if (z < 2) {
    __bf16* op = qkv + (size_t)z * (size_t)(Bz * Tz) * Hz;
#pragma unroll
    for (int mi = 0; mi < 2; ++mi) {
      const size_t rbase = row0 + wm * 32 + mi * 16 + 8 * hl;
#pragma unroll
      for (int ni = 0; ni < 4; ++ni) {
        const int c = col0 + wn * 64 + ni * 16 + ln;
#pragma unroll
        for (int j = 0; j < 8; ++j)
          op[(rbase + j) * Hz + c] = (__bf16)acc[mi][ni][j];
      }
    }
  } else {
    __bf16* op = qkv + 2 * (size_t)(Bz * Tz) * Hz;
    const size_t bbat = row0 / Tz;        // 128-row blocks never straddle a batch
    const int    t0   = (int)(row0 % Tz);
#pragma unroll
    for (int mi = 0; mi < 2; ++mi) {
      const int tb = t0 + wm * 32 + mi * 16 + 8 * hl;
#pragma unroll
      for (int ni = 0; ni < 4; ++ni) {
        const int c = col0 + wn * 64 + ni * 16 + ln;
        __bf16* row = op + ((size_t)bbat * Hz + c) * Tz + tb;
#pragma unroll
        for (int j = 0; j < 8; ++j) row[j] = (__bf16)acc[mi][ni][j];
      }
    }
  }
}

// ---------------------------------------------------------------------------
// Kernel 2: causal flash attention (no 1/sqrt(d) scaling, per reference).
// Grid: (T/16, B). Block: 256 threads = 8 waves.
// Block owns 16 query rows; wave w owns h-slice [w*128, w*128+128).
// Per 32-key step:
//   S partials (8 WMMA/wave, kT B-frags straight from global b128 loads)
//   -> deterministic LDS tree reduction -> wave0 online softmax -> P (bf16)
//   -> O^T += V^T * P^T (8 WMMA/wave, V^T A-frags straight from global).
// ---------------------------------------------------------------------------
__global__ __launch_bounds__(256) void attn_kernel(
    const __bf16* __restrict__ q,
    const __bf16* __restrict__ k,
    const __bf16* __restrict__ vT,
    float* __restrict__ out) {
  __shared__ float  Spart[8][512];   // per-wave S partials [m*32 + key]
  __shared__ float  Sred[512];       // reduced scores
  __shared__ __bf16 Pbuf[16 * 32];   // softmax probs, row-major [m][key]
  __shared__ float  rscale[16];      // exp(m_old - m_new) per row
  __shared__ float  rsum[16];        // final softmax denominators

  const int tid  = threadIdx.x;
  const int lane = tid & 31;
  const int wid  = tid >> 5;
  const int ln   = lane & 15;
  const int hl   = lane >> 4;
  const int bb   = blockIdx.y;
  const int q0   = blockIdx.x * 16;
  const int hw   = wid * 128;

  const float NEG_INF = -3.0e38f;

  // Preload q fragments for this wave's h-slice (A-frags, kept in VGPRs).
  v16bf qf[4];
#pragma unroll
  for (int c = 0; c < 4; ++c) {
    const __bf16* p = q + ((size_t)(bb * Tz + q0 + ln)) * Hz + hw + c * 32;
    qf[c] = frag_ld(p + 8 * hl, p + 16 + 8 * hl);
  }

  v8f o[8] = {};                       // O^T accum: frag f covers h = hw+f*16..+15
  float run_m = NEG_INF, run_s = 0.f;  // wave0 lanes<16 only

  for (int s0 = 0; s0 < q0 + 16; s0 += 32) {
    // ---- S partials over this wave's 128-wide h slice
    v8f s[2] = {};
#pragma unroll
    for (int c = 0; c < 4; ++c) {
#pragma unroll
      for (int kh = 0; kh < 2; ++kh) {
        const __bf16* kp = k + ((size_t)(bb * Tz + s0 + kh * 16 + ln)) * Hz
                             + hw + c * 32 + 8 * hl;
        v16bf bf = frag_ld(kp, kp + 16);   // B-frag of k^T: lane=key, elem=h
        s[kh] = WMMA_BF16(qf[c], bf, s[kh]);
      }
    }
#pragma unroll
    for (int kh = 0; kh < 2; ++kh)
#pragma unroll
      for (int j = 0; j < 8; ++j)
        Spart[wid][(j + 8 * hl) * 32 + kh * 16 + ln] = s[kh][j];
    __syncthreads();

    // ---- deterministic reduction over the 8 waves
    for (int e = tid; e < 512; e += 256) {
      float a = 0.f;
#pragma unroll
      for (int w = 0; w < 8; ++w) a += Spart[w][e];
      Sred[e] = a;
    }
    __syncthreads();

    // ---- online softmax (wave0, one row per lane)
    if (wid == 0 && lane < 16) {
      const int m = lane, qg = q0 + m;
      float mx = run_m;
      for (int j = 0; j < 32; ++j) {
        const float v = (s0 + j <= qg) ? Sred[m * 32 + j] : NEG_INF;
        mx = fmaxf(mx, v);
      }
      const float sc = __expf(run_m - mx);
      float sum = 0.f;
      for (int j = 0; j < 32; ++j) {
        float p = 0.f;
        if (s0 + j <= qg) p = __expf(Sred[m * 32 + j] - mx);
        sum += p;
        Pbuf[m * 32 + j] = (__bf16)p;
      }
      run_s = run_s * sc + sum;
      run_m = mx;
      rscale[m] = sc;
    }
    __syncthreads();

    // ---- O^T = O^T*scale + V^T @ P^T
    const float sc = rscale[ln];                   // per-lane: column = q row ln
    union { v16bf v; v4i d[2]; } pu;               // B-frag of P^T: lane=qrow, elem=key
    pu.d[0] = *(const v4i*)&Pbuf[ln * 32 + 8 * hl];
    pu.d[1] = *(const v4i*)&Pbuf[ln * 32 + 16 + 8 * hl];
#pragma unroll
    for (int f = 0; f < 8; ++f) {
#pragma unroll
      for (int j = 0; j < 8; ++j) o[f][j] *= sc;
      const __bf16* vp = vT + ((size_t)(bb * Hz + hw + f * 16 + ln)) * Tz + s0;
      v16bf va = frag_ld(vp + 8 * hl, vp + 16 + 8 * hl);  // A-frag of V^T
      o[f] = WMMA_BF16(va, pu.v, o[f]);
    }
    // (no barrier needed: Spart rewrite is fenced by this iteration's barriers)
  }

  if (wid == 0 && lane < 16) rsum[lane] = run_s;
  __syncthreads();

  // ---- write out: O^T element (h = hw + f*16 + j + 8*hl, row = q0 + ln)
  const float inv = 1.0f / rsum[ln];
#pragma unroll
  for (int f = 0; f < 8; ++f) {
    float* op = out + ((size_t)(bb * Tz + q0 + ln)) * Hz + hw + f * 16 + 8 * hl;
    v4f lo = { o[f][0] * inv, o[f][1] * inv, o[f][2] * inv, o[f][3] * inv };
    v4f hi = { o[f][4] * inv, o[f][5] * inv, o[f][6] * inv, o[f][7] * inv };
    *(v4f*)(op + 0) = lo;
    *(v4f*)(op + 4) = hi;
  }
}

// ---------------------------------------------------------------------------
extern "C" void kernel_launch(void* const* d_in, const int* in_sizes, int n_in,
                              void* d_out, int out_size, void* d_ws, size_t ws_size,
                              hipStream_t stream) {
  (void)in_sizes; (void)n_in; (void)out_size; (void)ws_size;
  const float* x  = (const float*)d_in[0];
  const float* Wq = (const float*)d_in[1];
  const float* Wk = (const float*)d_in[2];
  const float* Wv = (const float*)d_in[3];
  float* out = (float*)d_out;

  // Workspace (bf16): [ q | k | vT | xb | Wt ]  ~= 262 MB
  __bf16* ws = (__bf16*)d_ws;
  const size_t plane = (size_t)Bz * Tz * Hz;   // 33.5M halves
  __bf16* qkv = ws;                            // 3 planes
  __bf16* xb  = ws + 3 * plane;                // B*T*D
  __bf16* Wt  = xb + (size_t)Bz * Tz * Dz;     // 3*H*D

  convert_x_kernel<<<(Bz * Tz * Dz) / 2048, 256, 0, stream>>>(x, xb);
  transpose_w_kernel<<<dim3(Hz / 256, Dz, 3), 256, 0, stream>>>(Wq, Wk, Wv, Wt);
  qkv_gemm_kernel<<<dim3(Hz / 128, (Bz * Tz) / 128, 3), 256, 0, stream>>>(xb, Wt, qkv);
  attn_kernel<<<dim3(Tz / 16, Bz), 256, 0, stream>>>(qkv, qkv + plane,
                                                     qkv + 2 * plane, out);
}
